// TimesNetMSU_2362232012864
// MI455X (gfx1250) — matrix operand order repaired and checked by reading, not compiled, not run on hardware
//
#include <hip/hip_runtime.h>
#include <hip/hip_bf16.h>

typedef __attribute__((ext_vector_type(16))) _Float16     v16h;
typedef __attribute__((ext_vector_type(8)))  float        v8f;
typedef __attribute__((ext_vector_type(4)))  unsigned int v4u;
typedef __attribute__((ext_vector_type(8)))  int          v8i;
typedef __attribute__((ext_vector_type(4)))  int          v4i;

#ifndef __has_builtin
#define __has_builtin(x) 0
#endif
#if __has_builtin(__builtin_amdgcn_tensor_load_to_lds) && \
    __has_builtin(__builtin_amdgcn_s_wait_tensorcnt)
#define HAS_TDM 1
#else
#define HAS_TDM 0
#endif

#define BB_   256
#define TT_   128
#define FIN_  10
#define DD_   64
#define DFF_  256
#define LMAX_ 176      // max L over all periods is 168; round to 11*16
#define NTAP_ 121      // merged 11x11 kernel
#define MT_   (LMAX_/16)

// ---------------------------------------------------------------- utilities
__global__ void zero_f32_kernel(float* p, int n) {
  int i = blockIdx.x * blockDim.x + threadIdx.x;
  if (i < n) p[i] = 0.0f;
}

// ---------------------------------------------------------------- embedding
__global__ void embed_kernel(const float* __restrict__ X,
                             const float* __restrict__ w,
                             const float* __restrict__ bias,
                             float* __restrict__ xout) {
  int id = blockIdx.x * blockDim.x + threadIdx.x;
  if (id >= BB_ * TT_ * DD_) return;
  int h  = id % DD_;
  int bt = id / DD_;
  const float* xp = X + (size_t)bt * FIN_;
  const float* wp = w + (size_t)h * FIN_;
  float acc = bias[h];
#pragma unroll
  for (int f = 0; f < FIN_; ++f) acc += xp[f] * wp[f];
  xout[(size_t)bt * DD_ + h] = acc;
}

// ---------------------------------------------------------------- DFT pieces
__global__ void twiddle_kernel(float* __restrict__ tw) {
  int i = blockIdx.x * blockDim.x + threadIdx.x;  // i = f*T + t
  if (i >= 65 * TT_) return;
  int f = i / TT_, t = i % TT_;
  float th = 6.2831853071795864769f * (float)f * (float)t / (float)TT_;
  tw[2 * i + 0] = __cosf(th);
  tw[2 * i + 1] = __sinf(th);
}

__global__ void spectrum_kernel(const float* __restrict__ x,
                                const float* __restrict__ tw,
                                float* __restrict__ ampBD) {
  int id = blockIdx.x * blockDim.x + threadIdx.x;
  if (id >= BB_ * 65 * DD_) return;
  int d = id % DD_;
  int f = (id / DD_) % 65;
  int b = id / (DD_ * 65);
  const float* xp = x + (size_t)b * TT_ * DD_ + d;
  const float* tp = tw + 2 * (size_t)f * TT_;
  float re = 0.f, im = 0.f;
  for (int t = 0; t < TT_; ++t) {
    float xv = xp[(size_t)t * DD_];
    re += xv * tp[2 * t + 0];
    im += xv * tp[2 * t + 1];
  }
  atomicAdd(&ampBD[b * 65 + f], sqrtf(re * re + im * im) * (1.0f / 64.0f));
}

__global__ void ampf_kernel(const float* __restrict__ ampBD, float* __restrict__ ampF) {
  int f = blockIdx.x * blockDim.x + threadIdx.x;
  if (f >= 65) return;
  float s = 0.f;
  for (int b = 0; b < BB_; ++b) s += ampBD[b * 65 + f];
  ampF[f] = (f == 0) ? 0.0f : s * (1.0f / (float)BB_);
}

// binfo: [k*3+0]=p, [k*3+1]=H, [k*3+2]=L, [9+k]=freq index
__global__ void topk_kernel(const float* __restrict__ ampF, int* __restrict__ binfo) {
  if (threadIdx.x != 0 || blockIdx.x != 0) return;
  float v[65];
  for (int f = 0; f < 65; ++f) v[f] = ampF[f];
  v[0] = -1.0f;
  for (int k = 0; k < 3; ++k) {
    int best = 1; float bv = v[1];
    for (int f = 2; f < 65; ++f) if (v[f] > bv) { bv = v[f]; best = f; }
    v[best] = -1.0f;
    int p = TT_ / best;
    int H = (TT_ + p - 1) / p;
    binfo[k * 3 + 0] = p;
    binfo[k * 3 + 1] = H;
    binfo[k * 3 + 2] = H * p;
    binfo[9 + k] = best;
  }
}

__global__ void wsoft_kernel(const float* __restrict__ ampBD,
                             const int* __restrict__ binfo,
                             float* __restrict__ wsoft) {
  int b = blockIdx.x * blockDim.x + threadIdx.x;
  if (b >= BB_) return;
  float a0 = ampBD[b * 65 + binfo[9]];
  float a1 = ampBD[b * 65 + binfo[10]];
  float a2 = ampBD[b * 65 + binfo[11]];
  float mx = fmaxf(a0, fmaxf(a1, a2));
  float e0 = __expf(a0 - mx), e1 = __expf(a1 - mx), e2 = __expf(a2 - mx);
  float inv = 1.0f / (e0 + e1 + e2);
  wsoft[b * 3 + 0] = e0 * inv;
  wsoft[b * 3 + 1] = e1 * inv;
  wsoft[b * 3 + 2] = e2 * inv;
}

__global__ void xprep_kernel(const float* __restrict__ xc,
                             _Float16* __restrict__ xf,
                             float* __restrict__ xn) {
  int id = blockIdx.x * blockDim.x + threadIdx.x;
  if (id >= BB_ * LMAX_ * DD_) return;
  int d = id % DD_;
  int t = (id / DD_) % LMAX_;
  int b = id / (DD_ * LMAX_);
  float v = 0.0f;
  if (t < TT_) {
    v = xc[((size_t)b * TT_ + t) * DD_ + d];
    xn[((size_t)b * TT_ + t) * DD_ + d] = v;
  }
  xf[((size_t)b * LMAX_ + t) * DD_ + d] = (_Float16)v;
}

// ------------------------------------------------------- merged-weight prep
// Merge 6 centered kernels (1,3,..,11) into one 11x11 kernel and swizzle into
// WMMA B-fragment order: wfrag[otile][tap][ichunk][lane][16 halves], where
// lane<16 holds (N=lane, K=0..15) and lane>=16 holds (N=lane-16, K=16..31).
__global__ void prepw_kernel(const float* w0, const float* w1, const float* w2,
                             const float* w3, const float* w4, const float* w5,
                             _Float16* __restrict__ wfrag, int O, int I) {
  int id = blockIdx.x * blockDim.x + threadIdx.x;
  if (id >= O * I * NTAP_) return;
  int tap = id % NTAP_;
  int i   = (id / NTAP_) % I;
  int o   = id / (NTAP_ * I);
  int dh = tap / 11 - 5, dw = tap % 11 - 5;
  int amax = max(abs(dh), abs(dw));
  float s = 0.0f;
#pragma unroll
  for (int j = 0; j < 6; ++j) {
    if (j >= amax) {
      int kk = 2 * j + 1;
      const float* wp = (j == 0) ? w0 : (j == 1) ? w1 : (j == 2) ? w2
                        : (j == 3) ? w3 : (j == 4) ? w4 : w5;
      s += wp[((size_t)(o * I + i)) * (kk * kk) + (dh + j) * kk + (dw + j)];
    }
  }
  int icch = i >> 5;
  int K    = i & 31;
  int n    = o & 15;
  int lane = n + ((K & 16) ? 16 : 0);
  int icchs = I >> 5;
  size_t idx = (((size_t)(o >> 4) * NTAP_ + tap) * icchs + icch) * 512
               + (size_t)lane * 16 + (K & 15);
  wfrag[idx] = (_Float16)s;
}

__global__ void prepb_kernel(const float* b0, const float* b1, const float* b2,
                             const float* b3, const float* b4, const float* b5,
                             float* __restrict__ bc, int O) {
  int o = blockIdx.x * blockDim.x + threadIdx.x;
  if (o >= O) return;
  bc[o] = b0[o] + b1[o] + b2[o] + b3[o] + b4[o] + b5[o];
}

// ------------------------------------------------------- WMMA implicit GEMM
// One WG = 4 wave32; each wave owns one 16-out-channel tile; all waves share a
// per-tap A tile (16 spatial rows x IC channels) staged into LDS in WMMA
// A-fragment order (1 barrier / tap).  B panels (2KB, pre-swizzled fragments)
// are DMA'd into LDS with the Tensor Data Mover, double-buffered one step
// ahead and synchronized with s_wait_tensorcnt.
// MODE 0: epilogue = GELU((acc+bc)/6) -> f16 act buffer (conv1)
// MODE 1: epilogue = xnext += wsoft[b,branch] * (acc+bc)/6  for t<T (conv2)
template <int IC, int ICCH, int MODE>
__global__ void __launch_bounds__(128)
conv_wmma_kernel(const _Float16* __restrict__ xin,
                 const _Float16* __restrict__ wfrag,
                 const float* __restrict__ bc,
                 const int* __restrict__ binfo, int branch, int OC,
                 _Float16* __restrict__ out16, float* __restrict__ xnext,
                 const float* __restrict__ wsoft) {
  constexpr int NICP  = ICCH / 2;          // B-panel steps per tap (1 or 4)
  constexpr int NSTEP = NTAP_ * NICP;
  __shared__ __align__(32) _Float16 As[2][16 * IC];   // per-tap A tile
  __shared__ __align__(32) _Float16 Bs[2][4][1024];   // per-wave B panels

  const int p = binfo[branch * 3 + 0];
  const int H = binfo[branch * 3 + 1];
  const int L = binfo[branch * 3 + 2];

  const int mtile = blockIdx.x;
  const int bb = mtile / MT_;
  const int t0 = (mtile % MT_) * 16;
  if (t0 >= L) return;  // block-uniform: EXEC stays all-ones for WMMA region

  const int tid   = threadIdx.x;
  const int wave  = tid >> 5;
  const int lane  = tid & 31;
  const int otile = blockIdx.y * 4 + wave;
  const int o0    = otile * 16;

  // staging role: thread -> (target lane sl, 4-half group sg)
  const int sl = tid >> 2;
  const int sg = tid & 3;
  const int sr = sl & 15;
  // A fragment K layout: lane<16 holds K {0..15}; lane>=16 {16..31}
  const int kst = sg * 4 + ((sl >= 16) ? 16 : 0);

  const int st = t0 + sr;
  const int rowvalid = (st < L);
  const int sh = st / p, sw = st % p;

#if HAS_TDM
  // Issue one TDM copy of a 2KB pre-swizzled weight panel into this wave's
  // LDS slot.  D# per CDNA5 ISA 8.3/8.4: count=1, type=2 ("image"),
  // data_size=8B, 1 x 256 tile, stride 256.  (6-arg builtin form: groups
  // 0..3 + extra SGPR group + cpol; unused groups zeroed.)
  auto issue_tdm = [&](int s2) {
    const int tap2 = s2 / NICP, icp2 = s2 % NICP;
    unsigned long long ga = (unsigned long long)(size_t)(wfrag +
        (((size_t)otile * NTAP_ + tap2) * ICCH + icp2 * 2) * 512);
    unsigned lds = (unsigned)(size_t)&Bs[s2 & 1][wave][0];
    v4u g0;
    g0.x = 1u;                                             // count=1
    g0.y = lds;                                            // lds_addr
    g0.z = (unsigned)ga;                                   // global_addr lo
    g0.w = ((unsigned)(ga >> 32) & 0x01FFFFFFu) | 0x80000000u;  // hi | type=2
    v8i g1;
    g1[0] = 3 << 16;        // data_size = 8B
    g1[1] = 256 << 16;      // tensor_dim0 = 256 (lo16 at bit48)
    g1[2] = 1 << 16;        // tensor_dim0 hi = 0, tensor_dim1 = 1
    g1[3] = 256 << 16;      // tile_dim0 = 256
    g1[4] = 1;              // tile_dim1 = 1
    g1[5] = 256;            // tensor_dim0_stride = 256
    g1[6] = 0;
    g1[7] = 0;
    v4i z4 = {0, 0, 0, 0};
    v8i z8 = {0, 0, 0, 0, 0, 0, 0, 0};
    __builtin_amdgcn_tensor_load_to_lds(g0, g1, z4, z4, z8, 0);
  };
  issue_tdm(0);
#endif

  v8f acc = {};
  int s = 0;
  for (int tap = 0; tap < NTAP_; ++tap) {
    // ---- stage full-IC A tile for this tap (zero-masked conv window) ----
    const int dh = tap / 11 - 5, dw = tap % 11 - 5;
    const int hs = sh + dh, ws = sw + dw;
    const bool tv = rowvalid && ((unsigned)hs < (unsigned)H) &&
                    ((unsigned)ws < (unsigned)p);
    const int tsrc = tv ? (hs * p + ws) : 0;
    const _Float16* src = xin + ((size_t)bb * LMAX_ + tsrc) * IC + kst;
#pragma unroll
    for (int ic = 0; ic < ICCH; ++ic) {
      uint2 v = *(const uint2*)(src + ic * 32);
      if (!tv) { v.x = 0u; v.y = 0u; }
      *(uint2*)(&As[tap & 1][ic * 512 + sl * 16 + sg * 4]) = v;
    }
    __syncthreads();
    // ---- consume: ICCH WMMAs, B panels double-buffered via TDM ----
#pragma unroll
    for (int icp = 0; icp < NICP; ++icp, ++s) {
#if HAS_TDM
      if (s + 1 < NSTEP) {
        issue_tdm(s + 1);
        __builtin_amdgcn_s_wait_tensorcnt(1);
      } else {
        __builtin_amdgcn_s_wait_tensorcnt(0);
      }
#endif
#pragma unroll
      for (int h2 = 0; h2 < 2; ++h2) {
        const int ic = icp * 2 + h2;
        v16h a = *(const v16h*)(&As[tap & 1][ic * 512 + lane * 16]);
#if HAS_TDM
        v16h bm = *(const v16h*)(&Bs[s & 1][wave][h2 * 512 + lane * 16]);
#else
        const _Float16* wp = wfrag +
            (((size_t)otile * NTAP_ + tap) * ICCH + ic) * 512 + lane * 16;
        v16h bm = *(const v16h*)wp;
#endif
        acc = __builtin_amdgcn_wmma_f32_16x16x32_f16(
            false, a, false, bm, (short)0, acc, false, false);
      }
    }
  }

  // ---- epilogue: lane l, vgpr v -> (m = v + 8*(l>=16), n = l&15) ----
  const int l16 = lane >> 4;
  const int n = lane & 15;
  const float wk = (MODE == 1) ? wsoft[bb * 3 + branch] : 0.0f;
#pragma unroll
  for (int v = 0; v < 8; ++v) {
    int m = v + 8 * l16;
    int t = t0 + m;
    if (t >= L) continue;
    int o = o0 + n;
    float val = (acc[v] + bc[o]) * (1.0f / 6.0f);
    if (MODE == 0) {
      float g = 0.5f * val * (1.0f + erff(val * 0.70710678118654752f));
      out16[((size_t)bb * LMAX_ + t) * OC + o] = (_Float16)g;
    } else {
      if (t < TT_) {
        float* px = xnext + ((size_t)bb * TT_ + t) * DD_ + o;
        *px += wk * val;
      }
    }
  }
}

// ---------------------------------------------------------------- head
__global__ void score_kernel(const float* __restrict__ x,
                             const float* __restrict__ a1w,
                             const float* __restrict__ a1b,
                             const float* __restrict__ a2w,
                             const float* __restrict__ a2b,
                             float* __restrict__ score) {
  int id = blockIdx.x * blockDim.x + threadIdx.x;
  if (id >= BB_ * TT_) return;
  const float* xp = x + (size_t)id * DD_;
  float s = a2b[0];
  for (int g = 0; g < DD_; ++g) {
    float acc = a1b[g];
    const float* wp = a1w + (size_t)g * DD_;
    for (int d = 0; d < DD_; ++d) acc += wp[d] * xp[d];
    s += a2w[g] * tanhf(acc);
  }
  score[id] = s;
}

__global__ void pool_kernel(const float* __restrict__ x,
                            const float* __restrict__ score,
                            float* __restrict__ ae) {
  __shared__ float sm[TT_];
  __shared__ float red[TT_];
  int b = blockIdx.x, t = threadIdx.x;
  float s = score[b * TT_ + t];
  red[t] = s; __syncthreads();
  for (int off = 64; off; off >>= 1) {
    if (t < off) red[t] = fmaxf(red[t], red[t + off]);
    __syncthreads();
  }
  float mx = red[0]; __syncthreads();
  float e = __expf(s - mx);
  sm[t] = e; red[t] = e; __syncthreads();
  for (int off = 64; off; off >>= 1) {
    if (t < off) red[t] += red[t + off];
    __syncthreads();
  }
  float inv = 1.0f / red[0];
  if (t < DD_) {
    float acc = 0.f;
    for (int tt = 0; tt < TT_; ++tt)
      acc += sm[tt] * x[((size_t)b * TT_ + tt) * DD_ + t];
    ae[b * DD_ + t] = acc * inv;
  }
}

__global__ void head_kernel(const float* __restrict__ ae,
                            const float* __restrict__ l1w, const float* __restrict__ l1b,
                            const float* __restrict__ bng, const float* __restrict__ bnb,
                            const float* __restrict__ l2w, const float* __restrict__ l2b,
                            float* __restrict__ out) {
  __shared__ float hs[DD_];
  int b = blockIdx.x, g = threadIdx.x;
  float acc = l1b[g];
  const float* aep = ae + (size_t)b * DD_;
  for (int d = 0; d < DD_; ++d) acc += l1w[g * DD_ + d] * aep[d];
  acc = acc * rsqrtf(1.0f + 1e-5f) * bng[g] + bnb[g];
  hs[g] = fmaxf(acc, 0.0f);
  __syncthreads();
  if (g < 2) {
    float o = l2b[g];
    for (int d = 0; d < DD_; ++d) o += l2w[g * DD_ + d] * hs[d];
    out[b * 2 + g] = o;
  }
}

// ---------------------------------------------------------------- launcher
extern "C" void kernel_launch(void* const* d_in, const int* in_sizes, int n_in,
                              void* d_out, int out_size, void* d_ws, size_t ws_size,
                              hipStream_t stream) {
  if (n_in < 61) return;
  const float* X      = (const float*)d_in[0];
  const float* emb_w  = (const float*)d_in[1];
  const float* emb_b  = (const float*)d_in[2];
  const float* attn1w = (const float*)d_in[51];
  const float* attn1b = (const float*)d_in[52];
  const float* attn2w = (const float*)d_in[53];
  const float* attn2b = (const float*)d_in[54];
  const float* lin1w  = (const float*)d_in[55];
  const float* lin1b  = (const float*)d_in[56];
  const float* bng    = (const float*)d_in[57];
  const float* bnb    = (const float*)d_in[58];
  const float* lin2w  = (const float*)d_in[59];
  const float* lin2b  = (const float*)d_in[60];

  char* cur = (char*)d_ws;
  auto alloc = [&](size_t bytes) {
    char* r = cur;
    cur += (bytes + 255) & ~(size_t)255;
    return r;
  };
  float*    xA     = (float*)alloc((size_t)BB_ * TT_ * DD_ * 4);
  float*    xBv    = (float*)alloc((size_t)BB_ * TT_ * DD_ * 4);
  _Float16* xf16   = (_Float16*)alloc((size_t)BB_ * LMAX_ * DD_ * 2);
  _Float16* act1   = (_Float16*)alloc((size_t)BB_ * LMAX_ * DFF_ * 2);
  _Float16* wfrag1 = (_Float16*)alloc((size_t)16 * NTAP_ * 2 * 512 * 2);
  _Float16* wfrag2 = (_Float16*)alloc((size_t)4 * NTAP_ * 8 * 512 * 2);
  float*    bc1    = (float*)alloc(DFF_ * 4);
  float*    bc2    = (float*)alloc(DD_ * 4);
  float*    tw     = (float*)alloc((size_t)65 * TT_ * 2 * 4);
  float*    ampBD  = (float*)alloc((size_t)BB_ * 65 * 4);
  float*    ampF   = (float*)alloc(65 * 4);
  int*      binfo  = (int*)alloc(16 * 4);
  float*    wsoftp = (float*)alloc((size_t)BB_ * 3 * 4);
  float*    scorep = (float*)alloc((size_t)BB_ * TT_ * 4);
  float*    aep    = (float*)alloc((size_t)BB_ * DD_ * 4);

  twiddle_kernel<<<(65 * TT_ + 255) / 256, 256, 0, stream>>>(tw);
  embed_kernel<<<(BB_ * TT_ * DD_ + 255) / 256, 256, 0, stream>>>(X, emb_w, emb_b, xA);

  float* xcur = xA;
  float* xnext = xBv;
  for (int lyr = 0; lyr < 2; ++lyr) {
    int base = 3 + lyr * 24;
    zero_f32_kernel<<<(BB_ * 65 + 255) / 256, 256, 0, stream>>>(ampBD, BB_ * 65);
    spectrum_kernel<<<(BB_ * 65 * DD_ + 255) / 256, 256, 0, stream>>>(xcur, tw, ampBD);
    ampf_kernel<<<1, 65, 0, stream>>>(ampBD, ampF);
    topk_kernel<<<1, 1, 0, stream>>>(ampF, binfo);
    wsoft_kernel<<<(BB_ + 63) / 64, 64, 0, stream>>>(ampBD, binfo, wsoftp);
    xprep_kernel<<<(BB_ * LMAX_ * DD_ + 255) / 256, 256, 0, stream>>>(xcur, xf16, xnext);

    prepw_kernel<<<(DFF_ * DD_ * NTAP_ + 255) / 256, 256, 0, stream>>>(
        (const float*)d_in[base + 0], (const float*)d_in[base + 1],
        (const float*)d_in[base + 2], (const float*)d_in[base + 3],
        (const float*)d_in[base + 4], (const float*)d_in[base + 5],
        wfrag1, DFF_, DD_);
    prepb_kernel<<<1, DFF_, 0, stream>>>(
        (const float*)d_in[base + 6], (const float*)d_in[base + 7],
        (const float*)d_in[base + 8], (const float*)d_in[base + 9],
        (const float*)d_in[base + 10], (const float*)d_in[base + 11], bc1, DFF_);
    prepw_kernel<<<(DD_ * DFF_ * NTAP_ + 255) / 256, 256, 0, stream>>>(
        (const float*)d_in[base + 12], (const float*)d_in[base + 13],
        (const float*)d_in[base + 14], (const float*)d_in[base + 15],
        (const float*)d_in[base + 16], (const float*)d_in[base + 17],
        wfrag2, DD_, DFF_);
    prepb_kernel<<<1, DD_, 0, stream>>>(
        (const float*)d_in[base + 18], (const float*)d_in[base + 19],
        (const float*)d_in[base + 20], (const float*)d_in[base + 21],
        (const float*)d_in[base + 22], (const float*)d_in[base + 23], bc2, DD_);

    for (int k = 0; k < 3; ++k) {
      dim3 g1(BB_ * MT_, DFF_ / 64);   // 2816 x 4, 4 waves/WG
      conv_wmma_kernel<DD_, 2, 0><<<g1, 128, 0, stream>>>(
          xf16, wfrag1, bc1, binfo, k, DFF_, act1, nullptr, nullptr);
      dim3 g2(BB_ * MT_, 1);           // 2816 x 1
      conv_wmma_kernel<DFF_, 8, 1><<<g2, 128, 0, stream>>>(
          act1, wfrag2, bc2, binfo, k, DD_, nullptr, xnext, wsoftp);
    }
    float* tmp = xcur; xcur = xnext; xnext = tmp;
  }

  score_kernel<<<(BB_ * TT_ + 255) / 256, 256, 0, stream>>>(
      xcur, attn1w, attn1b, attn2w, attn2b, scorep);
  pool_kernel<<<BB_, TT_, 0, stream>>>(xcur, scorep, aep);
  head_kernel<<<BB_, DD_, 0, stream>>>(aep, lin1w, lin1b, bng, bnb, lin2w, lin2b,
                                       (float*)d_out);
}
